// QuantizedLoRALinear_48576080118365
// MI455X (gfx1250) — matrix operand order repaired and checked by reading, not compiled, not run on hardware
//
#include <hip/hip_runtime.h>

typedef __attribute__((ext_vector_type(16))) _Float16 v16h;
typedef __attribute__((ext_vector_type(8)))  _Float16 v8h;
typedef __attribute__((ext_vector_type(8)))  float    v8f;
typedef __attribute__((ext_vector_type(4)))  float    v4f;
typedef __attribute__((ext_vector_type(4)))  unsigned int u32x4;
typedef __attribute__((ext_vector_type(8)))  unsigned int u32x8;

// ---------------------------------------------------------------------------
// Kernel 1: convert x (f32) -> f16, 8 elements/thread, b128 loads/stores
// ---------------------------------------------------------------------------
__global__ __launch_bounds__(256) void qlora_convert_x(
    const float* __restrict__ x, _Float16* __restrict__ xh, long n) {
  long i = ((long)blockIdx.x * 256 + threadIdx.x) * 8;
  if (i + 8 > n) return;
  v4f a = *(const v4f*)(x + i);
  v4f b = *(const v4f*)(x + i + 4);
  v8h h;
#pragma unroll
  for (int j = 0; j < 4; ++j) { h[j] = (_Float16)a[j]; h[4 + j] = (_Float16)b[j]; }
  *(v8h*)(xh + i) = h;
}

// ---------------------------------------------------------------------------
// Kernel 2: fold base weight + LoRA into f16 effective weight
//   W_eff[o,i] = (float)W_int[o,i]*scale + 2.0 * sum_r B[o,r]*A[r,i]
// ---------------------------------------------------------------------------
__global__ __launch_bounds__(256) void qlora_fold_w(
    const int* __restrict__ wint, const float* __restrict__ wscale,
    const float* __restrict__ A, const float* __restrict__ Bl,
    _Float16* __restrict__ wh, int K) {
  const int o = blockIdx.y;
  const int i = blockIdx.x * 2048 + threadIdx.x * 8;
  const float scale = wscale[0];
  const int* wp = wint + (size_t)o * K + i;
  float acc[8];
#pragma unroll
  for (int j = 0; j < 8; ++j) acc[j] = (float)wp[j] * scale;
  const float* brow = Bl + (size_t)o * 16;
#pragma unroll
  for (int r = 0; r < 16; ++r) {
    const float br = 2.0f * brow[r];   // SCALING = 32/16 = 2.0
    const float* ap = A + (size_t)r * K + i;
#pragma unroll
    for (int j = 0; j < 8; ++j) acc[j] = fmaf(br, ap[j], acc[j]);
  }
  v8h h;
#pragma unroll
  for (int j = 0; j < 8; ++j) h[j] = (_Float16)acc[j];
  *(v8h*)(wh + (size_t)o * K + i) = h;
}

// ---------------------------------------------------------------------------
// TDM: one instruction loads a (rows x 32-half) tile (row stride K halves)
// from global into LDS, hardware-padding each 64B row with 16B so the LDS
// row stride is 80B (40 halves) -> conflict-free ds_load_b128 fragments.
// ---------------------------------------------------------------------------
__device__ __forceinline__ void tdm_load_tile(unsigned lds_off,
                                              unsigned long long gaddr,
                                              unsigned K, unsigned rows) {
  u32x4 g0;
  g0[0] = 1u;                                     // count=1 (valid descriptor)
  g0[1] = lds_off;                                // LDS byte address
  g0[2] = (unsigned)gaddr;                        // global_addr[31:0]
  g0[3] = (unsigned)((gaddr >> 32) & 0x01ffffffu) // global_addr[56:32]
          | (2u << 30);                           // type=2 ("image")
  u32x8 g1;
  g1[0] = (1u << 16)      // data_size = 2 bytes
        | (1u << 20)      // pad_enable
        | (3u << 22)      // pad_interval: 16 DWORDs = 64B
        | (3u << 25);     // pad_amount:    4 DWORDs = 16B
  g1[1] = (K & 0xffffu) << 16;   // tensor_dim0[15:0] = K
  g1[2] = 0u;                    // tensor_dim0[31:16]=0, tensor_dim1[15:0]=0
  g1[3] = 1u | (32u << 16);      // tensor_dim1[31:16]=1 (dim1=65536), tile_dim0=32
  g1[4] = rows;                  // tile_dim1 = rows, tile_dim2=0
  g1[5] = K;                     // tensor_dim0_stride[31:0] = K
  g1[6] = 0u;
  g1[7] = 0u;
  asm volatile("tensor_load_to_lds %0, %1" :: "s"(g0), "s"(g1) : "memory");
}

// ---------------------------------------------------------------------------
// Kernel 3: WMMA GEMM  out[M,N] = X[M,K] * W[N,K]^T  (f16 in, f32 out)
// 256 threads = 8 waves (wave32). Block tile 128(M) x 256(N), K-step 32.
// Wave grid 2(M) x 4(N); each wave: 64x64 = 4x4 16x16 tiles -> 16 WMMA and
// 16 ds_load_b128 per K-step (B frags loaded once, reused for all 4 M-rows).
// TDM double-buffered LDS staging: wave0 streams A (128x32), wave1 B (256x32).
// ---------------------------------------------------------------------------
__global__ __launch_bounds__(256) void qlora_wmma_gemm(
    const _Float16* __restrict__ X, const _Float16* __restrict__ W,
    float* __restrict__ out, int M, int N, int K) {
  constexpr int LS = 40;  // padded row stride in halves (80B), set by TDM pad
  __shared__ __align__(16) _Float16 As[2][128 * LS];
  __shared__ __align__(16) _Float16 Bs[2][256 * LS];

  const int tid  = threadIdx.x;
  const int lane = tid & 31;
  const int wave = tid >> 5;
  const int wm   = wave & 1;   // wave M index (0..1), 64 rows each
  const int wn   = wave >> 1;  // wave N index (0..3), 64 cols each
  const size_t m0 = (size_t)blockIdx.y * 128;
  const size_t n0 = (size_t)blockIdx.x * 256;

  // Tile base addresses (global) and LDS buffer offsets.
  const unsigned long long gA =
      (unsigned long long)(uintptr_t)(X + m0 * (size_t)K);
  const unsigned long long gB =
      (unsigned long long)(uintptr_t)(W + n0 * (size_t)K);
  const unsigned ldsA0 = (unsigned)(uintptr_t)&As[0][0];
  const unsigned ldsA1 = (unsigned)(uintptr_t)&As[1][0];
  const unsigned ldsB0 = (unsigned)(uintptr_t)&Bs[0][0];
  const unsigned ldsB1 = (unsigned)(uintptr_t)&Bs[1][0];

  // Prologue: stage K-slice 0 into buffer 0 (one TDM issue per matrix).
  if (wave == 0)      tdm_load_tile(ldsA0, gA, (unsigned)K, 128u);
  else if (wave == 1) tdm_load_tile(ldsB0, gB, (unsigned)K, 256u);

  v8f acc[4][4] = {};

  // fragment addressing (ISA 7.12.2 layouts, wave32)
  const int lrow = lane & 15;
  const int hi16 = lane >> 4;
  const int kba  = hi16 * 8;   // A: lanes 16-31 start at K=8 (and K=24)
  const int kbb  = hi16 * 8;   // B: lanes 16-31 start at K=8 (and K=24)

  const int nk = K / 32;
  for (int kt = 0; kt < nk; ++kt) {
    const int cur = kt & 1;
    // Waves 0/1 wait for their outstanding tensor DMA; barrier publishes LDS.
    if (wave < 2) __builtin_amdgcn_s_wait_tensorcnt(0);
    __syncthreads();

    // Issue next K-slice into the other buffer, overlapped with WMMA below.
    if (kt + 1 < nk) {
      const unsigned long long go = (unsigned long long)(kt + 1) * 64ull; // 32 halves
      if (wave == 0)
        tdm_load_tile(cur ? ldsA0 : ldsA1, gA + go, (unsigned)K, 128u);
      else if (wave == 1)
        tdm_load_tile(cur ? ldsB0 : ldsB1, gB + go, (unsigned)K, 256u);
    }

    const _Float16* Ab = As[cur];
    const _Float16* Bb = Bs[cur];

    v16h bf[4];
#pragma unroll
    for (int ni = 0; ni < 4; ++ni) {
      const _Float16* p = &Bb[(wn * 64 + ni * 16 + lrow) * LS + kbb];
      v8h lo = *(const v8h*)p;        // K kbb..kbb+7
      v8h hh = *(const v8h*)(p + 16); // K kbb+16..kbb+23
      bf[ni] = __builtin_shufflevector(lo, hh, 0, 1, 2, 3, 4, 5, 6, 7, 8, 9,
                                       10, 11, 12, 13, 14, 15);
    }
#pragma unroll
    for (int mi = 0; mi < 4; ++mi) {
      const _Float16* p = &Ab[(wm * 64 + mi * 16 + lrow) * LS + kba];
      v8h lo = *(const v8h*)p;         // K kba..kba+7
      v8h hh = *(const v8h*)(p + 16);  // K kba+16..kba+23
      v16h af = __builtin_shufflevector(lo, hh, 0, 1, 2, 3, 4, 5, 6, 7, 8, 9,
                                        10, 11, 12, 13, 14, 15);
#pragma unroll
      for (int ni = 0; ni < 4; ++ni)
        acc[mi][ni] = __builtin_amdgcn_wmma_f32_16x16x32_f16(
            false, af, false, bf[ni], (short)0, acc[mi][ni], false, false);
    }
  }

  // C/D layout: VGPR j -> M = j + 8*(lane>=16), N = lane&15
  const size_t rb = m0 + wm * 64 + hi16 * 8;
  const size_t cb = n0 + wn * 64 + lrow;
#pragma unroll
  for (int mi = 0; mi < 4; ++mi)
#pragma unroll
    for (int ni = 0; ni < 4; ++ni)
#pragma unroll
      for (int j = 0; j < 8; ++j)
        out[(rb + mi * 16 + j) * (size_t)N + cb + ni * 16] = acc[mi][ni][j];
}

// ---------------------------------------------------------------------------
extern "C" void kernel_launch(void* const* d_in, const int* in_sizes, int n_in,
                              void* d_out, int out_size, void* d_ws,
                              size_t ws_size, hipStream_t stream) {
  const float* x      = (const float*)d_in[0];   // [B,S,D_IN] f32
  const int*   wint   = (const int*)d_in[1];     // [D_OUT,D_IN] i32
  const float* wscale = (const float*)d_in[2];   // [1] f32
  const float* lA     = (const float*)d_in[3];   // [R,D_IN] f32
  const float* lB     = (const float*)d_in[4];   // [D_OUT,R] f32
  float*       out    = (float*)d_out;

  const int D_IN  = 4096;
  const int D_OUT = in_sizes[4] / 16;            // lora_B is [D_OUT, R=16]
  const long xn   = in_sizes[0];                 // B*S*D_IN
  const int  M    = (int)(xn / D_IN);            // 8192

  // workspace layout: x_f16 (xn halves) | w_eff_f16 (D_OUT*D_IN halves)
  _Float16* xh = (_Float16*)d_ws;
  _Float16* wh = xh + xn;

  qlora_convert_x<<<(xn / 8 + 255) / 256, 256, 0, stream>>>(x, xh, xn);
  qlora_fold_w<<<dim3(D_IN / 2048, D_OUT), 256, 0, stream>>>(wint, wscale, lA,
                                                             lB, wh, D_IN);
  qlora_wmma_gemm<<<dim3(D_OUT / 256, M / 128), 256, 0, stream>>>(
      xh, wh, out, M, D_OUT, D_IN);
}
